// AttentionSelfAttention_21990232555762
// MI455X (gfx1250) — compile-verified, matrix operand order checked by reading, not run on hardware
//
#include <hip/hip_runtime.h>
#include <hip/hip_bf16.h>

#define B_  8
#define D_  64
#define NH2 8          // 2*NH heads total
#define S_  1024
#define NCOL 1536      // q(512) | ka(256) | va(256) | ksa(256) | vsa(256)

typedef __attribute__((ext_vector_type(16))) __bf16 v16bf;
typedef __attribute__((ext_vector_type(8)))  __bf16 v8bf;
typedef __attribute__((ext_vector_type(8)))  float  v8f;

union V16 { v16bf v; v8bf h[2]; };

#define WMMA_BF16(A, Bf, C) __builtin_amdgcn_wmma_f32_16x16x32_bf16( \
    false, (A), false, (Bf), (short)0, (C), false, false)

// ---------------------------------------------------------------------------
// Kernel 0: pack weights for WMMA B-fragments.
//   Wt_all (1536 x 64) bf16 : row j = column j of [Wq|Wka|Wva|Wksa|Wvsa]
//   bias_all (1536) fp32
//   WoT (64 x 512) bf16     : row n = column n of Wo
// ---------------------------------------------------------------------------
__global__ __launch_bounds__(256) void prep_kernel(
    const float* __restrict__ Wq,  const float* __restrict__ bq,
    const float* __restrict__ Wka, const float* __restrict__ bka,
    const float* __restrict__ Wva, const float* __restrict__ bva,
    const float* __restrict__ Wksa,const float* __restrict__ bksa,
    const float* __restrict__ Wvsa,const float* __restrict__ bvsa,
    const float* __restrict__ Wo,
    __bf16* __restrict__ Wt_all, float* __restrict__ bias_all,
    __bf16* __restrict__ WoT)
{
    int idx = blockIdx.x * 256 + threadIdx.x;
    if (idx < NCOL * 64) {
        int j = idx >> 6, k = idx & 63;
        float v;
        if      (j < 512)  v = Wq  [k * 512 + j];
        else if (j < 768)  v = Wka [k * 256 + (j - 512)];
        else if (j < 1024) v = Wva [k * 256 + (j - 768)];
        else if (j < 1280) v = Wksa[k * 256 + (j - 1024)];
        else               v = Wvsa[k * 256 + (j - 1280)];
        Wt_all[idx] = (__bf16)v;
    } else if (idx < NCOL * 64 + NCOL) {
        int j = idx - NCOL * 64;
        float v;
        if      (j < 512)  v = bq  [j];
        else if (j < 768)  v = bka [j - 512];
        else if (j < 1024) v = bva [j - 768];
        else if (j < 1280) v = bksa[j - 1024];
        else               v = bvsa[j - 1280];
        bias_all[j] = v;
    } else if (idx < NCOL * 64 + NCOL + 64 * 512) {
        int i = idx - (NCOL * 64 + NCOL);
        int n = i >> 9, c = i & 511;
        WoT[i] = (__bf16)Wo[c * 64 + n];
    }
}

// ---------------------------------------------------------------------------
// Kernel 1: projections via WMMA. Block = 8 waves handles 16 rows x all 1536
// columns; wave w owns 12 n-tiles (192 columns). All per-tile control flow is
// wave-uniform (w via readfirstlane) -> scalar branches, no cndmask blends.
// ---------------------------------------------------------------------------
__global__ __launch_bounds__(256) void proj_kernel(
    const float* __restrict__ qx, const float* __restrict__ kvx,
    const float* __restrict__ pos,
    const __bf16* __restrict__ Wt_all, const float* __restrict__ bias_all,
    __bf16* __restrict__ qb, __bf16* __restrict__ kb, __bf16* __restrict__ vbT)
{
    const int s0   = blockIdx.x * 16;     // flat row = b*S + s (tile in-batch)
    const int b    = s0 >> 10;
    const int tid  = threadIdx.x;
    const int w    = __builtin_amdgcn_readfirstlane(tid >> 5);  // uniform
    const int lane = tid & 31;
    const int g    = lane >> 4;
    const int l    = lane & 15;

    __shared__ __bf16 xqb[16][64];
    __shared__ __bf16 xkb[16][64];

    for (int i = tid; i < 16 * 64; i += 256) {
        int m = i >> 6, d = i & 63;
        int grow = s0 + m;
        float p = pos[((grow) & (S_ - 1)) * 64 + d];
        xqb[m][d] = (__bf16)(qx[(size_t)grow * 64 + d]  + p);
        xkb[m][d] = (__bf16)(kvx[(size_t)grow * 64 + d] + p);
    }
    __syncthreads();

    v8f acc[12];
#pragma unroll
    for (int nt = 0; nt < 12; ++nt)
#pragma unroll
        for (int r = 0; r < 8; ++r) acc[nt][r] = 0.0f;

#pragma unroll
    for (int k0 = 0; k0 < 64; k0 += 32) {
        V16 aq, ak;
        const __bf16* aqp = &xqb[l][k0 + 8 * g];
        const __bf16* akp = &xkb[l][k0 + 8 * g];
        aq.h[0] = *(const v8bf*)(aqp);  aq.h[1] = *(const v8bf*)(aqp + 16);
        ak.h[0] = *(const v8bf*)(akp);  ak.h[1] = *(const v8bf*)(akp + 16);
#pragma unroll
        for (int nt = 0; nt < 12; ++nt) {
            int col_base = w * 192 + nt * 16;      // uniform (SGPR)
            V16 bfr;
            const __bf16* bp = Wt_all + (size_t)(col_base + l) * 64 + k0 + 8 * g;
            bfr.h[0] = *(const v8bf*)(bp);
            bfr.h[1] = *(const v8bf*)(bp + 16);
            if (col_base >= 512 && col_base < 1024) {
                acc[nt] = WMMA_BF16(ak.v, bfr.v, acc[nt]);
            } else {
                acc[nt] = WMMA_BF16(aq.v, bfr.v, acc[nt]);
            }
        }
    }

#pragma unroll
    for (int nt = 0; nt < 12; ++nt) {
        int col_base = w * 192 + nt * 16;          // uniform; tile never crosses
        float bias = bias_all[col_base + l];       // a 64-col head boundary
        if (col_base < 512) {
            int h = col_base >> 6, d = (col_base & 63) + l;
            __bf16* dst = qb + (((size_t)b * NH2 + h) * S_) * 64 + d;
#pragma unroll
            for (int r = 0; r < 8; ++r) {
                int s = (s0 + r + 8 * g) & (S_ - 1);
                dst[(size_t)s * 64] = (__bf16)(acc[nt][r] + bias);
            }
        } else if (col_base < 768) {
            int c = col_base - 512, h = c >> 6, d = (c & 63) + l;
            __bf16* dst = kb + (((size_t)b * NH2 + h) * S_) * 64 + d;
#pragma unroll
            for (int r = 0; r < 8; ++r) {
                int s = (s0 + r + 8 * g) & (S_ - 1);
                dst[(size_t)s * 64] = (__bf16)(acc[nt][r] + bias);
            }
        } else if (col_base < 1024) {
            int c = col_base - 768, h = c >> 6, d = (c & 63) + l;
            __bf16* dst = vbT + (((size_t)b * NH2 + h) * 64 + d) * S_;
#pragma unroll
            for (int r = 0; r < 8; ++r) {
                int s = (s0 + r + 8 * g) & (S_ - 1);
                dst[s] = (__bf16)(acc[nt][r] + bias);
            }
        } else if (col_base < 1280) {
            int c = col_base - 1024, h = (c >> 6) + 4, d = (c & 63) + l;
            __bf16* dst = kb + (((size_t)b * NH2 + h) * S_) * 64 + d;
#pragma unroll
            for (int r = 0; r < 8; ++r) {
                int s = (s0 + r + 8 * g) & (S_ - 1);
                dst[(size_t)s * 64] = (__bf16)(acc[nt][r] + bias);
            }
        } else {
            int c = col_base - 1280, h = (c >> 6) + 4, d = (c & 63) + l;
            __bf16* dst = vbT + (((size_t)b * NH2 + h) * 64 + d) * S_;
#pragma unroll
            for (int r = 0; r < 8; ++r) {
                int s = (s0 + r + 8 * g) & (S_ - 1);
                dst[s] = (__bf16)(acc[nt][r] + bias);
            }
        }
    }
}

// ---------------------------------------------------------------------------
// Kernel 2: fused attention. Block = 8 waves handles 32 query rows of one
// (b,h); each K/V B-fragment feeds two WMMAs (two 16-row A tiles).
// attn output stored non-temporally (write-once stream, keep L2 for K/V).
// ---------------------------------------------------------------------------
__global__ __launch_bounds__(256) void attn_kernel(
    const __bf16* __restrict__ qb, const __bf16* __restrict__ kb,
    const __bf16* __restrict__ vbT,
    float* __restrict__ attn_out, __bf16* __restrict__ ctxb)
{
    const int qt   = blockIdx.x;          // 0..31  (32-row query tiles)
    const int h    = blockIdx.y;
    const int b    = blockIdx.z;
    const int bh   = b * NH2 + h;
    const int q0   = qt * 32;
    const int tid  = threadIdx.x;
    const int w    = __builtin_amdgcn_readfirstlane(tid >> 5);  // uniform
    const int lane = tid & 31;
    const int g    = lane >> 4;
    const int l    = lane & 15;

    __shared__ __bf16 P[32][1040];
    __shared__ float  red[8][32];
    __shared__ float  ctxp[8][32][64];

    const float scale = 0.125f;

    // ---------------- GEMM1: S = Q @ K^T ------------------------------------
    v8f acc[2][8];
#pragma unroll
    for (int a = 0; a < 2; ++a)
#pragma unroll
        for (int t = 0; t < 8; ++t)
#pragma unroll
            for (int r = 0; r < 8; ++r) acc[a][t][r] = 0.0f;

    const __bf16* Qbase = qb + ((size_t)bh * S_ + q0) * 64;
    const __bf16* Kbase = kb + (size_t)bh * S_ * 64;

#pragma unroll
    for (int k0 = 0; k0 < 64; k0 += 32) {
        V16 afr[2];
#pragma unroll
        for (int a = 0; a < 2; ++a) {
            const __bf16* ap = Qbase + (size_t)(16 * a + l) * 64 + k0 + 8 * g;
            afr[a].h[0] = *(const v8bf*)(ap);
            afr[a].h[1] = *(const v8bf*)(ap + 16);
        }
#pragma unroll
        for (int t = 0; t < 8; ++t) {
            int n = w * 128 + t * 16 + l;
            V16 bfr;
            const __bf16* bp = Kbase + (size_t)n * 64 + k0 + 8 * g;
            bfr.h[0] = *(const v8bf*)(bp);
            bfr.h[1] = *(const v8bf*)(bp + 16);
            acc[0][t] = WMMA_BF16(afr[0].v, bfr.v, acc[0][t]);
            acc[1][t] = WMMA_BF16(afr[1].v, bfr.v, acc[1][t]);
        }
    }

    // ---------------- softmax over full 1024 columns ------------------------
    float rmax[2][8];
#pragma unroll
    for (int a = 0; a < 2; ++a)
#pragma unroll
        for (int r = 0; r < 8; ++r) rmax[a][r] = -3.0e38f;
#pragma unroll
    for (int a = 0; a < 2; ++a)
#pragma unroll
        for (int t = 0; t < 8; ++t)
#pragma unroll
            for (int r = 0; r < 8; ++r) {
                float v = acc[a][t][r] * scale;
                acc[a][t][r] = v;
                rmax[a][r] = fmaxf(rmax[a][r], v);
            }
#pragma unroll
    for (int a = 0; a < 2; ++a)
#pragma unroll
        for (int r = 0; r < 8; ++r)
            for (int off = 1; off < 16; off <<= 1)
                rmax[a][r] = fmaxf(rmax[a][r], __shfl_xor(rmax[a][r], off, 32));
    if (l == 0)
#pragma unroll
        for (int a = 0; a < 2; ++a)
#pragma unroll
            for (int r = 0; r < 8; ++r) red[w][16 * a + r + 8 * g] = rmax[a][r];
    __syncthreads();
    float gmax[2][8];
#pragma unroll
    for (int a = 0; a < 2; ++a)
#pragma unroll
        for (int r = 0; r < 8; ++r) {
            int m = 16 * a + r + 8 * g;
            float mx = red[0][m];
#pragma unroll
            for (int w2 = 1; w2 < 8; ++w2) mx = fmaxf(mx, red[w2][m]);
            gmax[a][r] = mx;
        }
    __syncthreads();

    float rsum[2][8];
#pragma unroll
    for (int a = 0; a < 2; ++a)
#pragma unroll
        for (int r = 0; r < 8; ++r) rsum[a][r] = 0.0f;
#pragma unroll
    for (int a = 0; a < 2; ++a)
#pragma unroll
        for (int t = 0; t < 8; ++t)
#pragma unroll
            for (int r = 0; r < 8; ++r) {
                float p = __expf(acc[a][t][r] - gmax[a][r]);
                acc[a][t][r] = p;
                rsum[a][r] += p;
            }
#pragma unroll
    for (int a = 0; a < 2; ++a)
#pragma unroll
        for (int r = 0; r < 8; ++r)
            for (int off = 1; off < 16; off <<= 1)
                rsum[a][r] += __shfl_xor(rsum[a][r], off, 32);
    if (l == 0)
#pragma unroll
        for (int a = 0; a < 2; ++a)
#pragma unroll
            for (int r = 0; r < 8; ++r) red[w][16 * a + r + 8 * g] = rsum[a][r];
    __syncthreads();
    float ginv[2][8];
#pragma unroll
    for (int a = 0; a < 2; ++a)
#pragma unroll
        for (int r = 0; r < 8; ++r) {
            int m = 16 * a + r + 8 * g;
            float s = red[0][m];
#pragma unroll
            for (int w2 = 1; w2 < 8; ++w2) s += red[w2][m];
            ginv[a][r] = 1.0f / s;
        }

    // write attn (fp32 output, non-temporal) + stash bf16 P tile in LDS
    float* attnBase = attn_out + (size_t)bh * S_ * S_ + (size_t)q0 * S_;
#pragma unroll
    for (int a = 0; a < 2; ++a)
#pragma unroll
        for (int t = 0; t < 8; ++t)
#pragma unroll
            for (int r = 0; r < 8; ++r) {
                float p = acc[a][t][r] * ginv[a][r];
                int m = 16 * a + r + 8 * g;
                int n = w * 128 + t * 16 + l;
                __builtin_nontemporal_store(p, attnBase + (size_t)m * S_ + n);
                P[m][n] = (__bf16)p;
            }
    __syncthreads();

    // ---------------- GEMM2: ctx = P @ V ------------------------------------
    v8f acc2[2][4];
#pragma unroll
    for (int a = 0; a < 2; ++a)
#pragma unroll
        for (int nt = 0; nt < 4; ++nt)
#pragma unroll
            for (int r = 0; r < 8; ++r) acc2[a][nt][r] = 0.0f;

#pragma unroll
    for (int kk = 0; kk < 4; ++kk) {
        int k0 = w * 128 + kk * 32;
        V16 afr[2];
#pragma unroll
        for (int a = 0; a < 2; ++a) {
            const __bf16* ap = &P[16 * a + l][k0 + 8 * g];
            afr[a].h[0] = *(const v8bf*)(ap);
            afr[a].h[1] = *(const v8bf*)(ap + 16);
        }
#pragma unroll
        for (int nt = 0; nt < 4; ++nt) {
            int n = nt * 16 + l;
            V16 bfr;
            const __bf16* bp = vbT + ((size_t)bh * 64 + n) * S_ + k0 + 8 * g;
            bfr.h[0] = *(const v8bf*)(bp);
            bfr.h[1] = *(const v8bf*)(bp + 16);
            acc2[0][nt] = WMMA_BF16(afr[0].v, bfr.v, acc2[0][nt]);
            acc2[1][nt] = WMMA_BF16(afr[1].v, bfr.v, acc2[1][nt]);
        }
    }
#pragma unroll
    for (int a = 0; a < 2; ++a)
#pragma unroll
        for (int nt = 0; nt < 4; ++nt)
#pragma unroll
            for (int r = 0; r < 8; ++r)
                ctxp[w][16 * a + r + 8 * g][nt * 16 + l] = acc2[a][nt][r];
    __syncthreads();

    // deterministic cross-wave reduction -> bf16 ctx (b, s, h*64+d)
    for (int i = tid; i < 32 * 64; i += 256) {
        int m = i >> 6, n = i & 63;
        float s = 0.0f;
#pragma unroll
        for (int w2 = 0; w2 < 8; ++w2) s += ctxp[w2][m][n];
        ctxb[((size_t)b * S_ + q0 + m) * (NH2 * 64) + h * 64 + n] = (__bf16)s;
    }
}

// ---------------------------------------------------------------------------
// Kernel 3: out = ctx @ Wo + bo via WMMA. Block = 8 waves handles 16 rows;
// wave w covers k-chunk [64w, 64w+64); serial LDS sum of the 8 waves' partials.
// ---------------------------------------------------------------------------
__global__ __launch_bounds__(256) void outproj_kernel(
    const __bf16* __restrict__ ctxb, const __bf16* __restrict__ WoT,
    const float* __restrict__ bo, float* __restrict__ out)
{
    const int s0   = blockIdx.x * 16;     // flat row b*S+s
    const int tid  = threadIdx.x;
    const int w    = __builtin_amdgcn_readfirstlane(tid >> 5);  // uniform
    const int lane = tid & 31;
    const int g    = lane >> 4;
    const int l    = lane & 15;

    __shared__ float part[8][16][64];

    v8f acc[4];
#pragma unroll
    for (int nt = 0; nt < 4; ++nt)
#pragma unroll
        for (int r = 0; r < 8; ++r) acc[nt][r] = 0.0f;

#pragma unroll
    for (int kk = 0; kk < 2; ++kk) {
        int k0 = w * 64 + kk * 32;
        V16 afr;
        const __bf16* ap = ctxb + (size_t)(s0 + l) * 512 + k0 + 8 * g;
        afr.h[0] = *(const v8bf*)(ap);
        afr.h[1] = *(const v8bf*)(ap + 16);
#pragma unroll
        for (int nt = 0; nt < 4; ++nt) {
            int n = nt * 16 + l;
            V16 bfr;
            const __bf16* bp = WoT + (size_t)n * 512 + k0 + 8 * g;
            bfr.h[0] = *(const v8bf*)(bp);
            bfr.h[1] = *(const v8bf*)(bp + 16);
            acc[nt] = WMMA_BF16(afr.v, bfr.v, acc[nt]);
        }
    }
#pragma unroll
    for (int nt = 0; nt < 4; ++nt)
#pragma unroll
        for (int r = 0; r < 8; ++r)
            part[w][r + 8 * g][nt * 16 + l] = acc[nt][r];
    __syncthreads();

    for (int i = tid; i < 16 * 64; i += 256) {
        int m = i >> 6, n = i & 63;
        float s = bo[n];
#pragma unroll
        for (int w2 = 0; w2 < 8; ++w2) s += part[w2][m][n];
        out[(size_t)(s0 + m) * 64 + n] = s;
    }
}

// ---------------------------------------------------------------------------
extern "C" void kernel_launch(void* const* d_in, const int* in_sizes, int n_in,
                              void* d_out, int out_size, void* d_ws, size_t ws_size,
                              hipStream_t stream)
{
    const float* kvx  = (const float*)d_in[0];
    const float* qx   = (const float*)d_in[1];
    const float* pos  = (const float*)d_in[2];
    const float* Wq   = (const float*)d_in[3];
    const float* bq   = (const float*)d_in[4];
    const float* Wka  = (const float*)d_in[5];
    const float* bka  = (const float*)d_in[6];
    const float* Wva  = (const float*)d_in[7];
    const float* bva  = (const float*)d_in[8];
    const float* Wksa = (const float*)d_in[9];
    const float* bksa = (const float*)d_in[10];
    const float* Wvsa = (const float*)d_in[11];
    const float* bvsa = (const float*)d_in[12];
    const float* Wo   = (const float*)d_in[13];
    const float* bo   = (const float*)d_in[14];

    float* out  = (float*)d_out;                          // (B,H,W,D)
    float* attn = out + (size_t)B_ * S_ * D_;             // (B, 8, S, S)

    char* ws = (char*)d_ws;
    const size_t nQKV = (size_t)B_ * NH2 * S_ * D_;       // 4,194,304 elems
    __bf16* qb    = (__bf16*)ws;  ws += nQKV * sizeof(__bf16);
    __bf16* kb    = (__bf16*)ws;  ws += nQKV * sizeof(__bf16);
    __bf16* vbT   = (__bf16*)ws;  ws += nQKV * sizeof(__bf16);
    __bf16* ctxb  = (__bf16*)ws;  ws += (size_t)B_ * S_ * 512 * sizeof(__bf16);
    __bf16* Wt_all= (__bf16*)ws;  ws += (size_t)NCOL * 64 * sizeof(__bf16);
    __bf16* WoT   = (__bf16*)ws;  ws += (size_t)64 * 512 * sizeof(__bf16);
    float* bias_all=(float*)ws;   ws += (size_t)NCOL * sizeof(float);

    const int prep_elems = NCOL * 64 + NCOL + 64 * 512;
    prep_kernel<<<(prep_elems + 255) / 256, 256, 0, stream>>>(
        Wq, bq, Wka, bka, Wva, bva, Wksa, bksa, Wvsa, bvsa, Wo,
        Wt_all, bias_all, WoT);

    proj_kernel<<<(B_ * S_) / 16, 256, 0, stream>>>(
        qx, kvx, pos, Wt_all, bias_all, qb, kb, vbT);

    dim3 g2(S_ / 32, NH2, B_);
    attn_kernel<<<g2, 256, 0, stream>>>(qb, kb, vbT, attn, ctxb);

    outproj_kernel<<<(B_ * S_) / 16, 256, 0, stream>>>(ctxb, WoT, bo, out);
}
